// GroupedQueryAttention_56848187130333
// MI455X (gfx1250) — matrix operand order repaired and checked by reading, not compiled, not run on hardware
//
#include <hip/hip_runtime.h>

typedef __bf16 bf16;
typedef __attribute__((ext_vector_type(16))) __bf16 v16bf;
typedef __attribute__((ext_vector_type(8)))  __bf16 v8bf;
typedef __attribute__((ext_vector_type(8)))  float  v8f;

#define B_    2
#define CTXN  2048
#define DIN   1024
#define DOUT  1024
#define NH    16
#define NKV   4
#define HD    64
#define GSZ   (NH / NKV)

// ---------------------------------------------------------------------------
// WMMA helpers (CDNA5 gfx1250: wave32, 16x16x32 bf16 -> f32)
// ---------------------------------------------------------------------------
__device__ __forceinline__ v8f wmma_bf16(v16bf a, v16bf b, v8f c) {
  return __builtin_amdgcn_wmma_f32_16x16x32_bf16(
      /*neg_a=*/false, a, /*neg_b=*/false, b,
      /*c_mod=*/(short)0, c, /*reuse_a=*/false, /*reuse_b=*/false);
}

// A-matrix 16x32 bf16 fragment: lane m = lane&15.
// element e in [0,16): K = (lane>>4)*16 + e (contiguous 16 halves)
// 'base' must already point at row_m*K + kbase + (lane>>4)*16.
__device__ __forceinline__ v16bf load_afrag(const bf16* __restrict__ base) {
  v8bf lo = *(const v8bf*)(base);
  v8bf hi = *(const v8bf*)(base + 8);
  v16bf a;
#pragma unroll
  for (int i = 0; i < 8; ++i) { a[i] = lo[i]; a[i + 8] = hi[i]; }
  return a;
}

// B-matrix 32x16 bf16 fragment: lane n = lane&15, K = (lane>>4)*16 + e.
// 'base' must already point at the 16 contiguous K values for this lane.
__device__ __forceinline__ v16bf load_bfrag(const bf16* __restrict__ base) {
  v8bf lo = *(const v8bf*)(base);
  v8bf hi = *(const v8bf*)(base + 8);
  v16bf b;
#pragma unroll
  for (int i = 0; i < 8; ++i) { b[i] = lo[i]; b[i + 8] = hi[i]; }
  return b;
}

__device__ __forceinline__ float redmax16(float v) {
#pragma unroll
  for (int off = 1; off < 16; off <<= 1) v = fmaxf(v, __shfl_xor(v, off, 32));
  return v;
}
__device__ __forceinline__ float redsum16(float v) {
#pragma unroll
  for (int off = 1; off < 16; off <<= 1) v += __shfl_xor(v, off, 32);
  return v;
}

// ---------------------------------------------------------------------------
// Prep kernels
// ---------------------------------------------------------------------------
__global__ void k_f32_to_bf16(const float* __restrict__ src, bf16* __restrict__ dst, int n) {
  int i = blockIdx.x * blockDim.x + threadIdx.x;
  if (i < n) dst[i] = (bf16)src[i];
}

// W fp32 [K][N] -> Wt bf16 [N][K]
__global__ void k_transpose_w(const float* __restrict__ W, bf16* __restrict__ Wt, int K, int N) {
  int i = blockIdx.x * blockDim.x + threadIdx.x;
  if (i >= K * N) return;
  int k = i / N, n = i - k * N;
  Wt[(size_t)n * K + k] = (bf16)W[i];
}

__global__ void k_rope_tables(float* __restrict__ cosT, float* __restrict__ sinT) {
  int i = blockIdx.x * blockDim.x + threadIdx.x;
  if (i >= CTXN * 32) return;
  int pos = i >> 5, d = i & 31;
  // inv_freq = 10000^{-d/32}
  float inv = expf(-(float)d * (logf(10000.0f) / 32.0f));
  float fr = (float)pos * inv;
  cosT[i] = cosf(fr);
  sinT[i] = sinf(fr);
}

// ---------------------------------------------------------------------------
// Register-blocked bf16 WMMA GEMM:
//   C[M,N](f32) = A[M,K](bf16,row-major) * Bt[N,K]^T
// One wave computes a 32(M) x 64(N) tile: 2 A-frags x 4 B-frags -> 8 WMMAs
// per 32-deep k-step from 12 b128 loads (1.5 loads/WMMA vs 4 for 16x16).
// ---------------------------------------------------------------------------
__global__ void __launch_bounds__(128)
k_gemm_bf16(const bf16* __restrict__ A, const bf16* __restrict__ Bt,
            float* __restrict__ C, int M, int N, int K) {
  int wave = (blockIdx.x * blockDim.x + threadIdx.x) >> 5;
  int lane = threadIdx.x & 31;
  int ntn = N >> 6;              // 64-wide column tiles
  int ntiles = (M >> 5) * ntn;   // 32-tall row tiles
  if (wave >= ntiles) return;
  int tn = wave % ntn;
  int tm = wave / ntn;
  int half = lane >> 4, l16 = lane & 15;

  const bf16* arow0 = A + (size_t)(tm * 32 + l16) * K + half * 16;
  const bf16* arow1 = arow0 + (size_t)16 * K;
  const bf16* brow  = Bt + (size_t)(tn * 64 + l16) * K + half * 16;

  v8f acc[2][4];
#pragma unroll
  for (int i = 0; i < 2; ++i)
#pragma unroll
    for (int j = 0; j < 4; ++j) acc[i][j] = (v8f){};

  for (int k0 = 0; k0 < K; k0 += 32) {
    v16bf a0 = load_afrag(arow0 + k0);
    v16bf a1 = load_afrag(arow1 + k0);
    v16bf b0 = load_bfrag(brow + k0);
    v16bf b1 = load_bfrag(brow + (size_t)16 * K + k0);
    v16bf b2 = load_bfrag(brow + (size_t)32 * K + k0);
    v16bf b3 = load_bfrag(brow + (size_t)48 * K + k0);
    acc[0][0] = wmma_bf16(a0, b0, acc[0][0]);
    acc[0][1] = wmma_bf16(a0, b1, acc[0][1]);
    acc[0][2] = wmma_bf16(a0, b2, acc[0][2]);
    acc[0][3] = wmma_bf16(a0, b3, acc[0][3]);
    acc[1][0] = wmma_bf16(a1, b0, acc[1][0]);
    acc[1][1] = wmma_bf16(a1, b1, acc[1][1]);
    acc[1][2] = wmma_bf16(a1, b2, acc[1][2]);
    acc[1][3] = wmma_bf16(a1, b3, acc[1][3]);
  }

  // C layout per 16x16 frag: lane row = r + (lane>>4)*8, col = lane&15
#pragma unroll
  for (int mi = 0; mi < 2; ++mi) {
    float* crow = C + (size_t)(tm * 32 + mi * 16 + half * 8) * N + tn * 64 + l16;
#pragma unroll
    for (int f = 0; f < 4; ++f) {
#pragma unroll
      for (int r = 0; r < 8; ++r) crow[(size_t)r * N + f * 16] = acc[mi][f][r];
    }
  }
}

// ---------------------------------------------------------------------------
// RoPE + repack kernels
// ---------------------------------------------------------------------------
// Qf fp32 [b*CTX][1024] -> Qb bf16 [b][h][pos][64] (RoPE applied)
__global__ void k_rope_q(const float* __restrict__ Qf, const float* __restrict__ cosT,
                         const float* __restrict__ sinT, bf16* __restrict__ Qb) {
  int i = blockIdx.x * blockDim.x + threadIdx.x;
  if (i >= B_ * NH * CTXN * 32) return;
  int d = i & 31;
  int pos = (i >> 5) & (CTXN - 1);
  int h = (i >> 16) & (NH - 1);
  int b = i >> 20;
  const float* row = Qf + ((size_t)(b * CTXN + pos)) * DOUT + h * HD;
  float x1 = row[d], x2 = row[d + 32];
  float c = cosT[pos * 32 + d], s = sinT[pos * 32 + d];
  size_t o = (((size_t)b * NH + h) * CTXN + pos) * HD;
  Qb[o + d] = (bf16)(x1 * c - x2 * s);
  Qb[o + d + 32] = (bf16)(x1 * s + x2 * c);
}

// Kf fp32 [b*CTX][256] -> Kb bf16 [b][kv][pos][64] (RoPE applied)
__global__ void k_rope_k(const float* __restrict__ Kf, const float* __restrict__ cosT,
                         const float* __restrict__ sinT, bf16* __restrict__ Kb) {
  int i = blockIdx.x * blockDim.x + threadIdx.x;
  if (i >= B_ * NKV * CTXN * 32) return;
  int d = i & 31;
  int pos = (i >> 5) & (CTXN - 1);
  int kv = (i >> 16) & (NKV - 1);
  int b = i >> 18;
  const float* row = Kf + ((size_t)(b * CTXN + pos)) * (NKV * HD) + kv * HD;
  float x1 = row[d], x2 = row[d + 32];
  float c = cosT[pos * 32 + d], s = sinT[pos * 32 + d];
  size_t o = (((size_t)b * NKV + kv) * CTXN + pos) * HD;
  Kb[o + d] = (bf16)(x1 * c - x2 * s);
  Kb[o + d + 32] = (bf16)(x1 * s + x2 * c);
}

// Vf fp32 [b*CTX][256] -> Vt bf16 [b][kv][d][pos]  (transposed for B-fragments)
__global__ void k_transpose_v(const float* __restrict__ Vf, bf16* __restrict__ Vt) {
  int i = blockIdx.x * blockDim.x + threadIdx.x;
  if (i >= B_ * NKV * CTXN * HD) return;
  int d = i & 63;
  int pos = (i >> 6) & (CTXN - 1);
  int kv = (i >> 17) & (NKV - 1);
  int b = i >> 19;
  float v = Vf[((size_t)(b * CTXN + pos)) * (NKV * HD) + kv * HD + d];
  Vt[(((size_t)(b * NKV + kv)) * HD + d) * CTXN + pos] = (bf16)v;
}

// ---------------------------------------------------------------------------
// Flash-style causal GQA attention.
// One wave per (b, kv, g, 16-query tile). Keys processed 32 at a time:
//   4 score WMMAs (Q 16x64 @ K^T 64x32) + 4 PV WMMAs (P 16x32 @ V 32x64).
// Online softmax state (row max / row sum) kept per C-layout row via shfl.
// P C-layout -> A-layout fragment transpose goes through LDS.
// ---------------------------------------------------------------------------
__global__ void __launch_bounds__(256)
k_attn(const bf16* __restrict__ Qb, const bf16* __restrict__ Kb,
       const bf16* __restrict__ Vt, bf16* __restrict__ ctxb) {
  __shared__ bf16 plds[8][16 * 32];  // per-wave 16x32 P tile (8 waves/block)

  int gwave = (blockIdx.x * blockDim.x + threadIdx.x) >> 5;
  int wl = threadIdx.x >> 5;
  int lane = threadIdx.x & 31;
  if (gwave >= B_ * NKV * GSZ * (CTXN / 16)) return;
  int qtile = gwave & (CTXN / 16 - 1);
  int g = (gwave >> 7) & (GSZ - 1);
  int kv = (gwave >> 9) & (NKV - 1);
  int b = gwave >> 11;
  int h = kv * GSZ + g;  // reference reshape (b,H,..) -> (b,KV,G,..): h = kv*G+g
  int half = lane >> 4, l16 = lane & 15;
  int qbase = qtile * 16;

  // Q fragments (A layout), row = query token
  const bf16* qptr = Qb + ((((size_t)b * NH + h) * CTXN + qbase + l16) * HD) + half * 16;
  v16bf qf0 = load_afrag(qptr);
  v16bf qf1 = load_afrag(qptr + 32);

  float mrow[8], lrow[8];
#pragma unroll
  for (int r = 0; r < 8; ++r) { mrow[r] = -__builtin_inff(); lrow[r] = 0.0f; }
  v8f O0 = {}, O1 = {}, O2 = {}, O3 = {};

  const bf16* kbaseptr = Kb + ((size_t)b * NKV + kv) * CTXN * HD;
  const bf16* vbaseptr = Vt + (((size_t)b * NKV + kv) * HD + l16) * CTXN;

  int qhi = qbase + 15;
  for (int j0 = 0; j0 <= qhi; j0 += 32) {
    // --- scores: S(16x32) = Q(16x64) @ K_tile^T ; B-frag lane n = key ---
    const bf16* kp0 = kbaseptr + (size_t)(j0 + l16) * HD + half * 16;
    const bf16* kp1 = kp0 + 16 * HD;
    v16bf bk00 = load_bfrag(kp0);       // keys j0..j0+15,  dims 0..31
    v16bf bk01 = load_bfrag(kp0 + 32);  // keys j0..j0+15,  dims 32..63
    v16bf bk10 = load_bfrag(kp1);       // keys j0+16..+31, dims 0..31
    v16bf bk11 = load_bfrag(kp1 + 32);
    v8f s0 = {}, s1 = {};
    s0 = wmma_bf16(qf0, bk00, s0);
    s0 = wmma_bf16(qf1, bk01, s0);
    s1 = wmma_bf16(qf0, bk10, s1);
    s1 = wmma_bf16(qf1, bk11, s1);

    int key0 = j0 + l16, key1 = j0 + 16 + l16;
#pragma unroll
    for (int r = 0; r < 8; ++r) {
      int qg = qbase + half * 8 + r;
      // reference scales scores by sqrt(HD)=8 before softmax; causal mask
      float a0 = (key0 <= qg) ? s0[r] * 8.0f : -__builtin_inff();
      float a1 = (key1 <= qg) ? s1[r] * 8.0f : -__builtin_inff();
      float vmax = redmax16(fmaxf(a0, a1));
      float mnew = fmaxf(mrow[r], vmax);
      float corr = __expf(mrow[r] - mnew);
      float p0 = __expf(a0 - mnew);
      float p1 = __expf(a1 - mnew);
      float rs = redsum16(p0 + p1);
      lrow[r] = lrow[r] * corr + rs;
      mrow[r] = mnew;
      O0[r] *= corr; O1[r] *= corr; O2[r] *= corr; O3[r] *= corr;
      // stash P tile (C layout) into LDS for cross-lane transpose
      plds[wl][(half * 8 + r) * 32 + l16] = (bf16)p0;
      plds[wl][(half * 8 + r) * 32 + 16 + l16] = (bf16)p1;
    }
    asm volatile("s_wait_dscnt 0x0" ::: "memory");

    // read P back in A-fragment layout: lane m = l16, K runs over 32 keys
    v16bf pa;
    {
      v8bf lo = *(const v8bf*)&plds[wl][l16 * 32 + half * 16];
      v8bf hi = *(const v8bf*)&plds[wl][l16 * 32 + half * 16 + 8];
#pragma unroll
      for (int i = 0; i < 8; ++i) { pa[i] = lo[i]; pa[i + 8] = hi[i]; }
    }

    // --- context accumulate: O(16x64) += P(16x32) @ V_tile(32x64) ---
    const bf16* vp = vbaseptr + j0 + half * 16;  // lane n = dim (f*16 + l16)
    O0 = wmma_bf16(pa, load_bfrag(vp + 0 * 16 * CTXN), O0);
    O1 = wmma_bf16(pa, load_bfrag(vp + 1 * 16 * CTXN), O1);
    O2 = wmma_bf16(pa, load_bfrag(vp + 2 * 16 * CTXN), O2);
    O3 = wmma_bf16(pa, load_bfrag(vp + 3 * 16 * CTXN), O3);
  }

  // epilogue: divide by softmax denominator, write ctx bf16 [token][1024]
#pragma unroll
  for (int r = 0; r < 8; ++r) {
    float inv = 1.0f / lrow[r];
    size_t token = (size_t)b * CTXN + qbase + half * 8 + r;
    bf16* crow = ctxb + token * DOUT + h * HD + l16;
    crow[0]  = (bf16)(O0[r] * inv);
    crow[16] = (bf16)(O1[r] * inv);
    crow[32] = (bf16)(O2[r] * inv);
    crow[48] = (bf16)(O3[r] * inv);
  }
}

// ---------------------------------------------------------------------------
// Host launch
// ---------------------------------------------------------------------------
static inline int cdiv(int a, int b) { return (a + b - 1) / b; }

extern "C" void kernel_launch(void* const* d_in, const int* in_sizes, int n_in,
                              void* d_out, int out_size, void* d_ws, size_t ws_size,
                              hipStream_t stream) {
  (void)in_sizes; (void)n_in; (void)out_size; (void)ws_size;
  const float* x  = (const float*)d_in[0];
  const float* Wq = (const float*)d_in[1];
  const float* Wk = (const float*)d_in[2];
  const float* Wv = (const float*)d_in[3];
  const float* Wo = (const float*)d_in[4];
  float* out = (float*)d_out;

  char* wsb = (char*)d_ws;
  size_t off = 0;
  auto take = [&](size_t bytes) -> void* {
    void* r = wsb + off;
    off += (bytes + 255) & ~(size_t)255;
    return r;
  };
  const size_t TOK = (size_t)B_ * CTXN;           // 4096
  bf16* xb   = (bf16*)take(TOK * DIN * 2);
  bf16* Wqt  = (bf16*)take((size_t)DOUT * DIN * 2);
  bf16* Wkt  = (bf16*)take((size_t)(NKV * HD) * DIN * 2);
  bf16* Wvt  = (bf16*)take((size_t)(NKV * HD) * DIN * 2);
  bf16* Wot  = (bf16*)take((size_t)DOUT * DOUT * 2);
  float* cosT = (float*)take((size_t)CTXN * 32 * 4);
  float* sinT = (float*)take((size_t)CTXN * 32 * 4);
  float* Qf  = (float*)take(TOK * DOUT * 4);
  float* Kf  = (float*)take(TOK * (NKV * HD) * 4);
  float* Vf  = (float*)take(TOK * (NKV * HD) * 4);
  bf16* Qb   = (bf16*)take(TOK * DOUT * 2);
  bf16* Kb   = (bf16*)take(TOK * (NKV * HD) * 2 + 8192);  // pad: causal tail reads
  bf16* Vt   = (bf16*)take(TOK * (NKV * HD) * 2 + 8192);
  bf16* ctxb = (bf16*)take(TOK * DOUT * 2);

  const int TPB = 256;

  // 1) bf16 conversions / transposes / rope tables
  {
    int n = (int)(TOK * DIN);
    k_f32_to_bf16<<<cdiv(n, TPB), TPB, 0, stream>>>(x, xb, n);
  }
  k_transpose_w<<<cdiv(DIN * DOUT, TPB), TPB, 0, stream>>>(Wq, Wqt, DIN, DOUT);
  k_transpose_w<<<cdiv(DIN * NKV * HD, TPB), TPB, 0, stream>>>(Wk, Wkt, DIN, NKV * HD);
  k_transpose_w<<<cdiv(DIN * NKV * HD, TPB), TPB, 0, stream>>>(Wv, Wvt, DIN, NKV * HD);
  k_transpose_w<<<cdiv(DOUT * DOUT, TPB), TPB, 0, stream>>>(Wo, Wot, DOUT, DOUT);
  k_rope_tables<<<cdiv(CTXN * 32, TPB), TPB, 0, stream>>>(cosT, sinT);

  // 2) QKV projections (bf16 WMMA, f32 out); 32x64 tile per wave
  auto gemm = [&](const bf16* A, const bf16* Bt, float* C, int M, int N, int K) {
    int waves = (M / 32) * (N / 64);
    int tpb = 128;  // 4 waves/block: relaxed VGPR budget for 8-accum blocking
    k_gemm_bf16<<<cdiv(waves * 32, tpb), tpb, 0, stream>>>(A, Bt, C, M, N, K);
  };
  gemm(xb, Wqt, Qf, (int)TOK, DOUT, DIN);
  gemm(xb, Wkt, Kf, (int)TOK, NKV * HD, DIN);
  gemm(xb, Wvt, Vf, (int)TOK, NKV * HD, DIN);

  // 3) RoPE + repack to WMMA-friendly bf16 layouts
  k_rope_q<<<cdiv(B_ * NH * CTXN * 32, TPB), TPB, 0, stream>>>(Qf, cosT, sinT, Qb);
  k_rope_k<<<cdiv(B_ * NKV * CTXN * 32, TPB), TPB, 0, stream>>>(Kf, cosT, sinT, Kb);
  k_transpose_v<<<cdiv(B_ * NKV * CTXN * HD, TPB), TPB, 0, stream>>>(Vf, Vt);

  // 4) causal GQA flash attention
  {
    int waves = B_ * NKV * GSZ * (CTXN / 16);  // 4096
    k_attn<<<cdiv(waves * 32, TPB), TPB, 0, stream>>>(Qb, Kb, Vt, ctxb);
  }

  // 5) output projection -> fp32 d_out
  gemm(ctxb, Wot, out, (int)TOK, DOUT, DOUT);
}
